// HLGLayers_61332132986999
// MI455X (gfx1250) — matrix operand order repaired and checked by reading, not compiled, not run on hardware
//
#include <hip/hip_runtime.h>
#include <hip/hip_bf16.h>
#include <math.h>

typedef __bf16 bf16;
typedef __attribute__((ext_vector_type(16))) __bf16 v16bf;
typedef __attribute__((ext_vector_type(8)))  __bf16 v8bf;
typedef __attribute__((ext_vector_type(8)))  float  v8f;

#define DIM   512
#define HEADS 16
#define HD    32
#define NTOK  49
#define MPAD  64
#define BATCH 2048

// ---------------------------------------------------------------- WMMA helpers

__device__ __forceinline__ v8f wmma_bf16(v16bf a, v16bf b, v8f c) {
  return __builtin_amdgcn_wmma_f32_16x16x32_bf16(false, a, false, b, (short)0, c, false, false);
}

__device__ __forceinline__ v8f zero8() {
  v8f z;
#pragma unroll
  for (int i = 0; i < 8; ++i) z[i] = 0.0f;
  return z;
}

__device__ __forceinline__ v16bf cat8(v8bf lo, v8bf hi) {
  return __builtin_shufflevector(lo, hi, 0, 1, 2, 3, 4, 5, 6, 7,
                                 8, 9, 10, 11, 12, 13, 14, 15);
}

// A fragment (16x32 bf16) from row-major A[M][K]: lane (hl = lane>>4, r = lane&15)
// holds row m0+r, K chunks {k0+hl*8 .. +7} and {k0+16+hl*8 .. +7}. No bounds.
__device__ __forceinline__ v16bf load_A_nb(const bf16* base, int ld, int m0, int k0, int lane) {
  int r = lane & 15, hl = lane >> 4;
  const bf16* p = base + (size_t)(m0 + r) * ld + k0 + hl * 8;
  v8bf lo = *(const v8bf*)p;
  v8bf hi = *(const v8bf*)(p + 16);
  return cat8(lo, hi);
}

// B fragment (32x16 bf16) where the K-major operand is stored transposed:
// X row-major [N][K], B[k][n] = X[n][k]. Lane holds col n0+(lane&15),
// K chunks {k0+hl*8 .. +7} and {k0+16+hl*8 .. +7}. No bounds.
__device__ __forceinline__ v16bf load_BT_nb(const bf16* X, int ldx, int k0, int n0, int lane) {
  int c = lane & 15, hl = lane >> 4;
  const bf16* p = X + (size_t)(n0 + c) * ldx + k0 + hl * 8;
  v8bf lo = *(const v8bf*)p;
  v8bf hi = *(const v8bf*)(p + 16);
  return cat8(lo, hi);
}

// ---------------------------------------------------------------- bias-table prep

__device__ void ln_relu_mm32(const float* in, const float* lw, const float* lb,
                             const float* w, const float* b, int ncol, float* out) {
  float m = 0.f;
#pragma unroll
  for (int i = 0; i < 32; ++i) m += in[i];
  m *= (1.f / 32.f);
  float v = 0.f;
#pragma unroll
  for (int i = 0; i < 32; ++i) { float d = in[i] - m; v += d * d; }
  v *= (1.f / 32.f);
  float rs = rsqrtf(v + 1e-5f);
  float t[32];
#pragma unroll
  for (int i = 0; i < 32; ++i) {
    float y = (in[i] - m) * rs * lw[i] + lb[i];
    t[i] = y > 0.f ? y : 0.f;
  }
  for (int o = 0; o < ncol; ++o) {
    float s = b[o];
    for (int i = 0; i < 32; ++i) s += t[i] * w[i * ncol + o];
    out[o] = s;
  }
}

__global__ void prep_bias_kernel(const float* pp_w, const float* pp_b,
                                 const float* p1lw, const float* p1lb,
                                 const float* p1w,  const float* p1b,
                                 const float* p2lw, const float* p2lb,
                                 const float* p2w,  const float* p2b,
                                 const float* p3lw, const float* p3lb,
                                 const float* p3w,  const float* p3b,
                                 float* biasTab) {
  __shared__ float pos_l[169 * HEADS];
  int tid = threadIdx.x;
  if (tid < 169) {
    float g0 = (float)(tid / 13 - 6);
    float g1 = (float)(tid % 13 - 6);
    float h0[32], h1[32], h2[32], h3[16];
    for (int o = 0; o < 32; ++o) h0[o] = g0 * pp_w[o] + g1 * pp_w[32 + o] + pp_b[o];
    ln_relu_mm32(h0, p1lw, p1lb, p1w, p1b, 32, h1);
    ln_relu_mm32(h1, p2lw, p2lb, p2w, p2b, 32, h2);
    ln_relu_mm32(h2, p3lw, p3lb, p3w, p3b, 16, h3);
    for (int h = 0; h < 16; ++h) pos_l[tid * 16 + h] = h3[h];
  }
  __syncthreads();
  for (int idx = tid; idx < HEADS * NTOK * NTOK; idx += blockDim.x) {
    int hh  = idx / (NTOK * NTOK);
    int rem = idx % (NTOK * NTOK);
    int i = rem / NTOK, j = rem % NTOK;
    int rel = ((i / 7 - j / 7) + 6) * 13 + ((i % 7 - j % 7) + 6);
    biasTab[idx] = pos_l[rel * 16 + hh];
  }
}

// -------------------------------------------------- weight convert (transpose)

// src: row-major [K][N] f32  ->  dstT: row-major [N][K] bf16
__global__ void cvt_bf16_T_kernel(const float* src, bf16* dstT, int K, int N) {
  int total = K * N;
  for (int idx = blockIdx.x * blockDim.x + threadIdx.x; idx < total;
       idx += gridDim.x * blockDim.x) {
    int n = idx / K, k = idx % K;
    dstT[idx] = (bf16)src[(size_t)k * N + n];
  }
}

// fused QKV weight, transposed: dstT[n][k], n in [0,1536)
__global__ void assemble_qkv_T_kernel(const float* qw, const float* kvw, bf16* dstT) {
  const int total = 1536 * DIM;
  for (int idx = blockIdx.x * blockDim.x + threadIdx.x; idx < total;
       idx += gridDim.x * blockDim.x) {
    int n = idx / DIM, k = idx % DIM;
    float v = (n < 512) ? qw[(size_t)k * 512 + n] : kvw[(size_t)k * 1024 + (n - 512)];
    dstT[idx] = (bf16)v;
  }
}

// ---------------------------------------------------------------- attention block
// grid = BATCH, 512 threads (16 waves). LDS: 4 x (64x512) bf16 + 16 x (16x64) bf16.

__global__ void __launch_bounds__(512) attn_kernel(
    const float* __restrict__ x, const float* __restrict__ ln1w,
    const float* __restrict__ ln1b, const float* __restrict__ q_b,
    const float* __restrict__ kv_b, const float* __restrict__ proj_b,
    const bf16* __restrict__ wqkvT, const bf16* __restrict__ wprojT,
    const float* __restrict__ biasTab, float* __restrict__ x1) {
  extern __shared__ char smem[];
  bf16* s_xn = (bf16*)smem;              // [64][512], later reused as attn output
  bf16* s_q  = s_xn + MPAD * DIM;        // [64][512]
  bf16* s_k  = s_q + MPAD * DIM;         // [64][512]  (token-major, d contiguous)
  bf16* s_vT = s_k + MPAD * DIM;         // [512][64]  (d-major, token contiguous)
  bf16* s_p  = s_vT + DIM * MPAD;        // 16 waves * 16x64 prob scratch

  const int b    = blockIdx.x;
  const int lane = threadIdx.x & 31;
  const int wave = threadIdx.x >> 5;
  const int hl   = lane >> 4;

  // ---- Phase 1: LN1 -> s_xn rows 0..48; zero rows 49..63
  for (int r = wave; r < NTOK; r += 16) {
    const float* rp = x + ((size_t)b * NTOK + r) * DIM + lane * 16;
    float v[16]; float s = 0.f;
#pragma unroll
    for (int j = 0; j < 16; ++j) { v[j] = rp[j]; s += v[j]; }
#pragma unroll
    for (int off = 16; off > 0; off >>= 1) s += __shfl_xor(s, off, 32);
    float mean = s * (1.f / DIM);
    float s2 = 0.f;
#pragma unroll
    for (int j = 0; j < 16; ++j) { float d = v[j] - mean; s2 += d * d; }
#pragma unroll
    for (int off = 16; off > 0; off >>= 1) s2 += __shfl_xor(s2, off, 32);
    float rstd = rsqrtf(s2 * (1.f / DIM) + 1e-5f);
    const float* wp = ln1w + lane * 16;
    const float* bp = ln1b + lane * 16;
    v8bf o0, o1;
#pragma unroll
    for (int j = 0; j < 8; ++j) o0[j] = (bf16)((v[j] - mean) * rstd * wp[j] + bp[j]);
#pragma unroll
    for (int j = 0; j < 8; ++j) o1[j] = (bf16)((v[8 + j] - mean) * rstd * wp[8 + j] + bp[8 + j]);
    bf16* dst = s_xn + r * DIM + lane * 16;
    *(v8bf*)dst = o0;
    *(v8bf*)(dst + 8) = o1;
  }
  {
    v8bf z;
#pragma unroll
    for (int j = 0; j < 8; ++j) z[j] = (bf16)0.0f;
    for (int r = NTOK + wave; r < MPAD; r += 16) {
      bf16* dst = s_xn + r * DIM + lane * 16;
      *(v8bf*)dst = z;
      *(v8bf*)(dst + 8) = z;
    }
  }
  __syncthreads();

  // ---- Phase 2: QKV GEMM (64x1536, K=512); q pre-scaled by 1/sqrt(HD)
  for (int t = wave; t < 4 * 96; t += 16) {
    int m0 = (t & 3) * 16;
    int n0 = (t >> 2) * 16;
    v8f acc = zero8();
    for (int k0 = 0; k0 < DIM; k0 += 32) {
      v16bf a  = load_A_nb(s_xn, DIM, m0, k0, lane);
      v16bf bb = load_BT_nb(wqkvT, DIM, k0, n0, lane);
      acc = wmma_bf16(a, bb, acc);
    }
    int col = n0 + (lane & 15);
    if (col < 512) {
      float bv = q_b[col];
#pragma unroll
      for (int i = 0; i < 8; ++i) {
        int row = m0 + i + 8 * hl;
        s_q[row * DIM + col] = (bf16)((acc[i] + bv) * 0.17677669529663687f);
      }
    } else if (col < 1024) {
      float bv = kv_b[col - 512];
      int dc = col - 512;
#pragma unroll
      for (int i = 0; i < 8; ++i) {
        int row = m0 + i + 8 * hl;
        s_k[row * DIM + dc] = (bf16)(acc[i] + bv);
      }
    } else {
      float bv = kv_b[col - 512];
      int dc = col - 1024;
#pragma unroll
      for (int i = 0; i < 8; ++i) {
        int row = m0 + i + 8 * hl;
        s_vT[dc * MPAD + row] = (bf16)(acc[i] + bv);   // transposed V
      }
    }
  }
  __syncthreads();

  // ---- Phase 3: per-head attention (wave == head)
  {
    const int h = wave, hc = h * HD;
    bf16* pbuf = s_p + wave * (16 * 64);
    const float* bt = biasTab + h * (NTOK * NTOK);
    for (int mt = 0; mt < 4; ++mt) {
      int m0 = mt * 16;
      v16bf aq = load_A_nb(s_q, DIM, m0, hc, lane);      // K = head dim (32)
      v8f st[4];
#pragma unroll
      for (int nt = 0; nt < 4; ++nt) {
        v16bf bk = load_BT_nb(s_k, DIM, hc, nt * 16, lane);  // B = K^T
        st[nt] = wmma_bf16(aq, bk, zero8());
      }
      // bias + row/col mask
#pragma unroll
      for (int nt = 0; nt < 4; ++nt) {
        int colb = nt * 16 + (lane & 15);
#pragma unroll
        for (int i = 0; i < 8; ++i) {
          int row = m0 + i + 8 * hl;
          if (row < NTOK && colb < NTOK) st[nt][i] += bt[row * NTOK + colb];
          else st[nt][i] = -1e30f;
        }
      }
      // softmax over 64 padded cols (rows in VGPR slots, cols = 4 frags x 16 lanes)
      float inv[8];
#pragma unroll
      for (int i = 0; i < 8; ++i) {
        float m1 = fmaxf(fmaxf(st[0][i], st[1][i]), fmaxf(st[2][i], st[3][i]));
#pragma unroll
        for (int off = 8; off > 0; off >>= 1) m1 = fmaxf(m1, __shfl_xor(m1, off, 32));
        float s = 0.f;
#pragma unroll
        for (int nt = 0; nt < 4; ++nt) { float e = __expf(st[nt][i] - m1); st[nt][i] = e; s += e; }
#pragma unroll
        for (int off = 8; off > 0; off >>= 1) s += __shfl_xor(s, off, 32);
        inv[i] = 1.f / s;
      }
#pragma unroll
      for (int nt = 0; nt < 4; ++nt)
#pragma unroll
        for (int i = 0; i < 8; ++i)
          pbuf[(i + 8 * hl) * 64 + nt * 16 + (lane & 15)] = (bf16)(st[nt][i] * inv[i]);

      // out tile = P @ V  (K = 64 padded tokens)
#pragma unroll
      for (int vt = 0; vt < 2; ++vt) {
        v8f acc = zero8();
#pragma unroll
        for (int kc = 0; kc < 2; ++kc) {
          v16bf ap = load_A_nb(pbuf, 64, 0, kc * 32, lane);
          v16bf bv = load_BT_nb(s_vT, MPAD, kc * 32, hc + vt * 16, lane);
          acc = wmma_bf16(ap, bv, acc);
        }
#pragma unroll
        for (int i = 0; i < 8; ++i) {
          int row = m0 + i + 8 * hl;
          s_xn[row * DIM + hc + vt * 16 + (lane & 15)] = (bf16)acc[i];
        }
      }
    }
  }
  __syncthreads();

  // ---- Phase 4: x1 = x + attn_out @ proj_w + proj_b
  for (int t = wave; t < 4 * 32; t += 16) {
    int m0 = (t & 3) * 16;
    int n0 = (t >> 2) * 16;
    v8f acc = zero8();
    for (int k0 = 0; k0 < DIM; k0 += 32) {
      v16bf a  = load_A_nb(s_xn, DIM, m0, k0, lane);
      v16bf bb = load_BT_nb(wprojT, DIM, k0, n0, lane);
      acc = wmma_bf16(a, bb, acc);
    }
    int col = n0 + (lane & 15);
    float pb = proj_b[col];
#pragma unroll
    for (int i = 0; i < 8; ++i) {
      int row = m0 + i + 8 * hl;
      if (row < NTOK) {
        size_t off = ((size_t)b * NTOK + row) * DIM + col;
        x1[off] = x[off] + acc[i] + pb;
      }
    }
  }
}

// ---------------------------------------------------------------- fused MLP
// grid = 100352/64 blocks, 512 threads; out = x1 + gelu(LN2(x1)@W1+b1)@W2+b2

__global__ void __launch_bounds__(512) mlp_kernel(
    const float* __restrict__ x1, const float* __restrict__ ln2w,
    const float* __restrict__ ln2b, const bf16* __restrict__ wfc1T,
    const float* __restrict__ fc1_b, const bf16* __restrict__ wfc2T,
    const float* __restrict__ fc2_b, float* __restrict__ out) {
  extern __shared__ char smem[];
  bf16* s_xn2 = (bf16*)smem;                 // 64 x 512
  bf16* s_h   = s_xn2 + 64 * DIM;            // 64 x 64 hidden chunk

  const int R0   = blockIdx.x * 64;
  const int lane = threadIdx.x & 31;
  const int wave = threadIdx.x >> 5;
  const int hl   = lane >> 4;

  // LN2 -> s_xn2 (bf16), lane owns 16 contiguous elements
  for (int r = wave; r < 64; r += 16) {
    const float* rp = x1 + (size_t)(R0 + r) * DIM + lane * 16;
    float v[16]; float s = 0.f;
#pragma unroll
    for (int j = 0; j < 16; ++j) { v[j] = rp[j]; s += v[j]; }
#pragma unroll
    for (int off = 16; off > 0; off >>= 1) s += __shfl_xor(s, off, 32);
    float mean = s * (1.f / DIM);
    float s2 = 0.f;
#pragma unroll
    for (int j = 0; j < 16; ++j) { float d = v[j] - mean; s2 += d * d; }
#pragma unroll
    for (int off = 16; off > 0; off >>= 1) s2 += __shfl_xor(s2, off, 32);
    float rstd = rsqrtf(s2 * (1.f / DIM) + 1e-5f);
    const float* wp = ln2w + lane * 16;
    const float* bp = ln2b + lane * 16;
    v8bf o0, o1;
#pragma unroll
    for (int j = 0; j < 8; ++j) o0[j] = (bf16)((v[j] - mean) * rstd * wp[j] + bp[j]);
#pragma unroll
    for (int j = 0; j < 8; ++j) o1[j] = (bf16)((v[8 + j] - mean) * rstd * wp[8 + j] + bp[8 + j]);
    bf16* dst = s_xn2 + r * DIM + lane * 16;
    *(v8bf*)dst = o0;
    *(v8bf*)(dst + 8) = o1;
  }
  __syncthreads();

  v8f acc[8];
#pragma unroll
  for (int tt = 0; tt < 8; ++tt) acc[tt] = zero8();

  for (int c0 = 0; c0 < 2048; c0 += 64) {
    // h chunk (64x64): 4x4 tiles, one per wave; K = 512
    int mt = wave & 3, nt = wave >> 2;
    v8f a1 = zero8();
    for (int k0 = 0; k0 < DIM; k0 += 32) {
      v16bf a  = load_A_nb(s_xn2, DIM, mt * 16, k0, lane);
      v16bf bb = load_BT_nb(wfc1T, DIM, k0, c0 + nt * 16, lane);
      a1 = wmma_bf16(a, bb, a1);
    }
    int colh = nt * 16 + (lane & 15);
    float b1 = fc1_b[c0 + colh];
#pragma unroll
    for (int i = 0; i < 8; ++i) {
      float v = a1[i] + b1;
      v = 0.5f * v * (1.f + erff(v * 0.70710678118654752f));   // exact GELU
      s_h[(mt * 16 + i + 8 * hl) * 64 + colh] = (bf16)v;
    }
    __syncthreads();
    // acc += h_chunk @ W2[c0:c0+64, :]; each wave owns 8 of 128 tiles
#pragma unroll
    for (int tt = 0; tt < 8; ++tt) {
      int t  = wave + 16 * tt;
      int m0 = (t & 3) * 16;
      int n0 = (t >> 2) * 16;
#pragma unroll
      for (int kc = 0; kc < 2; ++kc) {
        v16bf a  = load_A_nb(s_h, 64, m0, kc * 32, lane);
        v16bf bb = load_BT_nb(wfc2T, 2048, c0 + kc * 32, n0, lane);
        acc[tt] = wmma_bf16(a, bb, acc[tt]);
      }
    }
    __syncthreads();
  }

  // epilogue: residual + fc2 bias
#pragma unroll
  for (int tt = 0; tt < 8; ++tt) {
    int t  = wave + 16 * tt;
    int m0 = (t & 3) * 16;
    int n0 = (t >> 2) * 16;
    int col = n0 + (lane & 15);
    float b2 = fc2_b[col];
#pragma unroll
    for (int i = 0; i < 8; ++i) {
      size_t off = (size_t)(R0 + m0 + i + 8 * hl) * DIM + col;
      out[off] = x1[off] + acc[tt][i] + b2;
    }
  }
}

// ---------------------------------------------------------------- launcher

static constexpr size_t OFF_BIAS  = 0;                                // 153664 B
static constexpr size_t OFF_WQKV  = 154112;
static constexpr size_t OFF_WPROJ = OFF_WQKV  + (size_t)1536 * 512 * 2;
static constexpr size_t OFF_WFC1  = OFF_WPROJ + (size_t)512 * 512 * 2;
static constexpr size_t OFF_WFC2  = OFF_WFC1  + (size_t)2048 * 512 * 2;
static constexpr size_t OFF_X1    = OFF_WFC2  + (size_t)512 * 2048 * 2;

extern "C" void kernel_launch(void* const* d_in, const int* in_sizes, int n_in,
                              void* d_out, int out_size, void* d_ws, size_t ws_size,
                              hipStream_t stream) {
  (void)in_sizes; (void)n_in; (void)out_size; (void)ws_size;
  const float* x      = (const float*)d_in[0];
  const float* ln1_w  = (const float*)d_in[1];
  const float* ln1_b  = (const float*)d_in[2];
  const float* q_w    = (const float*)d_in[3];
  const float* q_b    = (const float*)d_in[4];
  const float* kv_w   = (const float*)d_in[5];
  const float* kv_b   = (const float*)d_in[6];
  const float* proj_w = (const float*)d_in[7];
  const float* proj_b = (const float*)d_in[8];
  const float* pp_w   = (const float*)d_in[9];
  const float* pp_b   = (const float*)d_in[10];
  const float* p1_ln_w = (const float*)d_in[11];
  const float* p1_ln_b = (const float*)d_in[12];
  const float* p1_w    = (const float*)d_in[13];
  const float* p1_b    = (const float*)d_in[14];
  const float* p2_ln_w = (const float*)d_in[15];
  const float* p2_ln_b = (const float*)d_in[16];
  const float* p2_w    = (const float*)d_in[17];
  const float* p2_b    = (const float*)d_in[18];
  const float* p3_ln_w = (const float*)d_in[19];
  const float* p3_ln_b = (const float*)d_in[20];
  const float* p3_w    = (const float*)d_in[21];
  const float* p3_b    = (const float*)d_in[22];
  const float* ln2_w   = (const float*)d_in[23];
  const float* ln2_b   = (const float*)d_in[24];
  const float* fc1_w   = (const float*)d_in[25];
  const float* fc1_b   = (const float*)d_in[26];
  const float* fc2_w   = (const float*)d_in[27];
  const float* fc2_b   = (const float*)d_in[28];

  char* ws = (char*)d_ws;
  float* biasTab = (float*)(ws + OFF_BIAS);
  bf16*  wqkvT   = (bf16*)(ws + OFF_WQKV);
  bf16*  wprojT  = (bf16*)(ws + OFF_WPROJ);
  bf16*  wfc1T   = (bf16*)(ws + OFF_WFC1);
  bf16*  wfc2T   = (bf16*)(ws + OFF_WFC2);
  float* x1      = (float*)(ws + OFF_X1);
  float* outp    = (float*)d_out;

  prep_bias_kernel<<<1, 256, 0, stream>>>(pp_w, pp_b, p1_ln_w, p1_ln_b, p1_w, p1_b,
                                          p2_ln_w, p2_ln_b, p2_w, p2_b,
                                          p3_ln_w, p3_ln_b, p3_w, p3_b, biasTab);
  assemble_qkv_T_kernel<<<1024, 256, 0, stream>>>(q_w, kv_w, wqkvT);
  cvt_bf16_T_kernel<<<512, 256, 0, stream>>>(proj_w, wprojT, 512, 512);    // -> [512][512]
  cvt_bf16_T_kernel<<<2048, 256, 0, stream>>>(fc1_w, wfc1T, 512, 2048);    // -> [2048][512]
  cvt_bf16_T_kernel<<<2048, 256, 0, stream>>>(fc2_w, wfc2T, 2048, 512);    // -> [512][2048]

  const size_t attn_lds = (size_t)(4 * MPAD * DIM + 16 * 16 * 64) * sizeof(bf16); // 294912
  attn_kernel<<<BATCH, 512, attn_lds, stream>>>(x, ln1_w, ln1_b, q_b, kv_b, proj_b,
                                                wqkvT, wprojT, biasTab, x1);

  const size_t mlp_lds = (size_t)(64 * DIM + 64 * 64) * sizeof(bf16);             // 73728
  mlp_kernel<<<(BATCH * NTOK) / 64, 512, mlp_lds, stream>>>(x1, ln2_w, ln2_b,
                                                            wfc1T, fc1_b, wfc2T, fc2_b, outp);
}
